// SingleStreamBlock_45844480918337
// MI455X (gfx1250) — matrix operand order repaired and checked by reading, not compiled, not run on hardware
//
#include <hip/hip_runtime.h>
#include <math.h>

// ---------------- shapes ----------------
#define L_    2048
#define H_    2048
#define NH_   16
#define HD_   128
#define MLP_  8192
#define QKV_  6144      // 3*H
#define N1_   14336     // 3*H + MLP
#define CATW_ 10240     // H + MLP

// ---------------- WMMA types & helpers ----------------
typedef __bf16 bf16x16 __attribute__((ext_vector_type(16)));
typedef float  v8f     __attribute__((ext_vector_type(8)));

union FragBF { bf16x16 v; uint4 q[2]; };

__device__ inline unsigned short f2bf(float f) {
    union { float f; unsigned u; } x; x.f = f;
    unsigned r = x.u + 0x7FFFu + ((x.u >> 16) & 1u);   // RNE
    return (unsigned short)(r >> 16);
}

// CDNA5 async DMA: global -> LDS, 16B per lane, tracked by ASYNCcnt.
// LDS byte offset: generic shared pointers carry the LDS offset in addr[31:0]
// (low 32 bits are the LDS address); the async op writes LDS at its base plus the VGPR offset.
__device__ inline void async_ld_b128(unsigned ldsoff, unsigned long long gaddr) {
    asm volatile("global_load_async_to_lds_b128 %0, %1, off"
                 :: "v"(ldsoff), "v"(gaddr)
                 : "memory");
}
#define WAIT_ASYNC_0() asm volatile("s_wait_asynccnt 0x0" ::: "memory")
#define WAIT_ASYNC_4() asm volatile("s_wait_asynccnt 0x4" ::: "memory")
#define WAIT_DS_0()    asm volatile("s_wait_dscnt 0x0"    ::: "memory")

// A-matrix 16x32 bf16 fragment from LDS (row-major, k contiguous).
// Lane layout (ISA 7.12.2): row = lane&15; lanes<16: K 0-7 & 16-23, lanes>=16: K 8-15 & 24-31.
__device__ inline bf16x16 frag_a_lds(const unsigned short* base, int ldk, int k0) {
    int lane = threadIdx.x & 31;
    int row = lane & 15, kh = lane >> 4;
    FragBF f;
    f.q[0] = *(const uint4*)(base + row * ldk + k0 + 8 * kh);
    f.q[1] = *(const uint4*)(base + row * ldk + k0 + 16 + 8 * kh);
    return f.v;
}
// B-matrix 32x16 bf16 fragment from LDS stored [n][k] row-major.
// Lane layout: col = lane&15; lanes<16: K 0-7 & 16-23, lanes>=16: K 8-15 & 24-31.
__device__ inline bf16x16 frag_b_lds(const unsigned short* base, int ldk, int colbase, int k0) {
    int lane = threadIdx.x & 31;
    int col = colbase + (lane & 15), kh = lane >> 4;
    const uint4* p = (const uint4*)(base + col * ldk + k0 + 8 * kh);
    FragBF f; f.q[0] = p[0]; f.q[1] = p[2];
    return f.v;
}
__device__ inline v8f wmma_bf16(bf16x16 a, bf16x16 b, v8f c) {
    return __builtin_amdgcn_wmma_f32_16x16x32_bf16(false, a, false, b, (short)0, c, false, false);
}

__device__ inline float gelu_tanh(float x) {
    return 0.5f * x * (1.0f + tanhf(0.7978845608028654f * (x + 0.044715f * x * x * x)));
}

// ---------------- kernel 0: f32 -> bf16 convert ----------------
__global__ void cvt_kernel(const float* __restrict__ src, unsigned short* __restrict__ dst, int n) {
    int i = blockIdx.x * blockDim.x + threadIdx.x;
    int stride = gridDim.x * blockDim.x;
    for (; i < n; i += stride) dst[i] = f2bf(src[i]);
}

// ---------------- kernel 1: mod = silu(vec) @ w_mod^T + b_mod ----------------
__global__ __launch_bounds__(256) void mod_kernel(const float* __restrict__ vec,
                                                  const float* __restrict__ wmod,
                                                  const float* __restrict__ bmod,
                                                  float* __restrict__ modout) {
    int j = blockIdx.x, t = threadIdx.x;
    float p = 0.f;
    for (int k = t; k < H_; k += 256) {
        float s = vec[k];
        p += (s / (1.f + __expf(-s))) * wmod[(size_t)j * H_ + k];
    }
    for (int m = 16; m >= 1; m >>= 1) p += __shfl_xor(p, m, 32);
    __shared__ float sm[8];
    if ((t & 31) == 0) sm[t >> 5] = p;
    __syncthreads();
    if (t == 0) {
        float s = 0.f;
        for (int i = 0; i < 8; i++) s += sm[i];
        modout[j] = s + bmod[j];
    }
}

// ---------------- kernel 2: LayerNorm + modulation -> bf16 ----------------
__global__ __launch_bounds__(256) void ln_kernel(const float* __restrict__ x,
                                                 const float* __restrict__ mod,
                                                 unsigned short* __restrict__ xmod) {
    __shared__ float s1[8], s2[8];
    int row = blockIdx.x, t = threadIdx.x;
    float sum = 0.f, ss = 0.f;
    for (int c = t; c < H_; c += 256) {
        float v = x[(size_t)row * H_ + c];
        sum += v; ss += v * v;
    }
    for (int m = 16; m >= 1; m >>= 1) { sum += __shfl_xor(sum, m, 32); ss += __shfl_xor(ss, m, 32); }
    if ((t & 31) == 0) { s1[t >> 5] = sum; s2[t >> 5] = ss; }
    __syncthreads();
    sum = 0.f; ss = 0.f;
    for (int i = 0; i < 8; i++) { sum += s1[i]; ss += s2[i]; }
    float mu = sum / (float)H_;
    float var = ss / (float)H_ - mu * mu;
    float rstd = rsqrtf(var + 1e-6f);
    for (int c = t; c < H_; c += 256) {
        float v = (x[(size_t)row * H_ + c] - mu) * rstd;
        float xm = (1.f + mod[H_ + c]) * v + mod[c];
        xmod[(size_t)row * H_ + c] = f2bf(xm);
    }
}

// ------- shared GEMM core: 128x128 block, BK=32, 8 waves (2x4), double-buffered async LDS ----
// ldsA/ldsB are each 2 * 128 * 40 bf16 (two buffers).
__device__ inline void gemm_core(const unsigned short* __restrict__ A,
                                 const unsigned short* __restrict__ B,
                                 int K, int rowBase, int colBase,
                                 unsigned short* ldsA, unsigned short* ldsB,
                                 v8f (&acc)[4][2]) {
    const int t = threadIdx.x;
    const int w = t >> 5, wm = w >> 2, wn = w & 3;
    const int i0 = t * 2,     r0 = i0 >> 2, c0 = i0 & 3;
    const int i1 = t * 2 + 1, r1 = i1 >> 2, c1 = i1 & 3;
    const unsigned baseA = (unsigned)(uintptr_t)ldsA;
    const unsigned baseB = (unsigned)(uintptr_t)ldsB;
    const int ldq = K >> 3;                      // row stride in uint4
    const int T = K >> 5;                        // number of 32-wide k tiles
    auto issue = [&](int kt, int buf) {
        unsigned boff = (unsigned)buf * (128u * 40u * 2u);
        const uint4* gA = (const uint4*)A + (size_t)rowBase * ldq + kt * 4;
        const uint4* gB = (const uint4*)B + (size_t)colBase * ldq + kt * 4;
        async_ld_b128(baseA + boff + (unsigned)(r0 * 40 + c0 * 8) * 2,
                      (unsigned long long)(uintptr_t)(gA + (size_t)r0 * ldq + c0));
        async_ld_b128(baseA + boff + (unsigned)(r1 * 40 + c1 * 8) * 2,
                      (unsigned long long)(uintptr_t)(gA + (size_t)r1 * ldq + c1));
        async_ld_b128(baseB + boff + (unsigned)(r0 * 40 + c0 * 8) * 2,
                      (unsigned long long)(uintptr_t)(gB + (size_t)r0 * ldq + c0));
        async_ld_b128(baseB + boff + (unsigned)(r1 * 40 + c1 * 8) * 2,
                      (unsigned long long)(uintptr_t)(gB + (size_t)r1 * ldq + c1));
    };
    issue(0, 0);
    for (int kt = 0; kt < T; kt++) {
        int buf = kt & 1;
        if (kt + 1 < T) {
            issue(kt + 1, buf ^ 1);     // prefetch next tile into other buffer
            WAIT_ASYNC_4();             // 4 newer per-wave transfers may remain in flight
        } else {
            WAIT_ASYNC_0();
        }
        __syncthreads();                // all waves' tile-kt data visible in LDS
        const unsigned short* lA = ldsA + buf * 128 * 40;
        const unsigned short* lB = ldsB + buf * 128 * 40;
        bf16x16 b0 = frag_b_lds(lB, 40, wn * 32, 0);
        bf16x16 b1 = frag_b_lds(lB, 40, wn * 32 + 16, 0);
        for (int mi = 0; mi < 4; mi++) {
            bf16x16 a = frag_a_lds(lA + (wm * 64 + mi * 16) * 40, 40, 0);
            acc[mi][0] = wmma_bf16(a, b0, acc[mi][0]);
            acc[mi][1] = wmma_bf16(a, b1, acc[mi][1]);
        }
        WAIT_DS_0();                    // frag reads retired before buffer is overwritten
        __syncthreads();
    }
}

// ---------------- kernel 3: ln1 = x_mod @ w1^T + b1; split qkv(f32) / gelu(mlp)->cat(bf16) ----
__global__ __launch_bounds__(256) void gemm1_kernel(const unsigned short* __restrict__ xmod,
                                                    const unsigned short* __restrict__ w1bf,
                                                    const float* __restrict__ b1,
                                                    float* __restrict__ qkv,
                                                    unsigned short* __restrict__ cat) {
    __shared__ alignas(16) unsigned short ldsA[2 * 128 * 40];
    __shared__ alignas(16) unsigned short ldsB[2 * 128 * 40];
    const int t = threadIdx.x, w = t >> 5, lane = t & 31;
    const int wm = w >> 2, wn = w & 3;
    const int rowBase = blockIdx.y * 128, colBase = blockIdx.x * 128;
    v8f acc[4][2]; v8f z = {};
    for (int a = 0; a < 4; a++) for (int b = 0; b < 2; b++) acc[a][b] = z;
    gemm_core(xmod, w1bf, H_, rowBase, colBase, ldsA, ldsB, acc);
    const int kh = lane >> 4, nl = lane & 15;
    for (int mi = 0; mi < 4; mi++)
        for (int ni = 0; ni < 2; ni++) {
            int colg = colBase + wn * 32 + ni * 16 + nl;
            float bias = b1[colg];
            for (int i = 0; i < 8; i++) {
                int rowg = rowBase + wm * 64 + mi * 16 + i + 8 * kh;
                float v = acc[mi][ni][i] + bias;
                if (colg < QKV_) qkv[(size_t)rowg * QKV_ + colg] = v;
                else cat[(size_t)rowg * CATW_ + H_ + (colg - QKV_)] = f2bf(gelu_tanh(v));
            }
        }
}

// ---------------- kernel 4: RMSNorm + RoPE on q/k; write q,k head-major bf16 and v^T ----------
__global__ __launch_bounds__(128) void rope_kernel(const float* __restrict__ qkv,
                                                   const float* __restrict__ pe,
                                                   const float* __restrict__ qs,
                                                   const float* __restrict__ ks,
                                                   unsigned short* __restrict__ qb,
                                                   unsigned short* __restrict__ kb,
                                                   unsigned short* __restrict__ vT) {
    __shared__ float smq[4], smk[4];
    __shared__ float rowq[HD_], rowk[HD_];
    int bid = blockIdx.x;
    int h = bid >> 11, pos = bid & (L_ - 1);
    int d = threadIdx.x;
    float qv = qkv[(size_t)pos * QKV_ + h * HD_ + d];
    float kv = qkv[(size_t)pos * QKV_ + H_ + h * HD_ + d];
    float vv = qkv[(size_t)pos * QKV_ + 2 * H_ + h * HD_ + d];
    float sq = qv * qv, sk = kv * kv;
    for (int m = 16; m >= 1; m >>= 1) { sq += __shfl_xor(sq, m, 32); sk += __shfl_xor(sk, m, 32); }
    if ((d & 31) == 0) { smq[d >> 5] = sq; smk[d >> 5] = sk; }
    __syncthreads();
    sq = smq[0] + smq[1] + smq[2] + smq[3];
    sk = smk[0] + smk[1] + smk[2] + smk[3];
    float qn = qv * rsqrtf(sq / (float)HD_ + 1e-6f) * qs[d];
    float kn = kv * rsqrtf(sk / (float)HD_ + 1e-6f) * ks[d];
    rowq[d] = qn; rowk[d] = kn;
    __syncthreads();
    int pair = d >> 1, r = d & 1;
    float qe, qo, ke, ko;
    if (r == 0) { qe = qn; qo = rowq[d + 1]; ke = kn; ko = rowk[d + 1]; }
    else        { qe = rowq[d - 1]; qo = qn; ke = rowk[d - 1]; ko = kn; }
    const float* pp = pe + ((size_t)pos * 64 + pair) * 4 + r * 2;
    float a = pp[0], b = pp[1];
    float qr = a * qe + b * qo;
    float kr = a * ke + b * ko;
    qb[(size_t)(h * L_ + pos) * HD_ + d] = f2bf(qr * 0.08838834764831845f); // 1/sqrt(128)
    kb[(size_t)(h * L_ + pos) * HD_ + d] = f2bf(kr);
    vT[((size_t)h * HD_ + d) * L_ + pos] = f2bf(vv);
}

// ---------------- kernel 5: flash attention (bf16 WMMA, online softmax, async tile DMA) ------
__global__ __launch_bounds__(128) void attn_kernel(const unsigned short* __restrict__ qb,
                                                   const unsigned short* __restrict__ kb,
                                                   const unsigned short* __restrict__ vT,
                                                   unsigned short* __restrict__ cat) {
    __shared__ alignas(16) unsigned short ldsQ[64 * 136];
    __shared__ alignas(16) unsigned short ldsK[32 * 136];
    __shared__ alignas(16) unsigned short ldsV[128 * 40];
    __shared__ alignas(16) unsigned short ldsP[4 * 16 * 40];
    const int t = threadIdx.x, w = t >> 5, lane = t & 31;
    const int h = blockIdx.y;
    const int q0 = blockIdx.x * 64;
    const unsigned baseQ = (unsigned)(uintptr_t)ldsQ;
    const unsigned baseK = (unsigned)(uintptr_t)ldsK;
    const unsigned baseV = (unsigned)(uintptr_t)ldsV;
    // stage Q tile 64x128 bf16 via async DMA
    for (int j = 0; j < 8; j++) {
        int i = j * 128 + t;
        int r = i >> 4, c4 = i & 15;
        async_ld_b128(baseQ + (unsigned)(r * 136 + c4 * 8) * 2,
                      (unsigned long long)(uintptr_t)((const uint4*)qb +
                          (size_t)(h * L_ + q0 + r) * 16 + c4));
    }
    WAIT_ASYNC_0();
    __syncthreads();
    bf16x16 qf[4];
    for (int f = 0; f < 4; f++) qf[f] = frag_a_lds(ldsQ + (w * 16) * 136, 136, f * 32);
    v8f acc[8]; v8f z = {};
    for (int n = 0; n < 8; n++) acc[n] = z;
    float m_[8], l_[8];
    for (int i = 0; i < 8; i++) { m_[i] = -INFINITY; l_[i] = 0.f; }

    for (int kt = 0; kt < L_ / 32; kt++) {
        WAIT_DS_0();                        // prior frag reads retired before overwrite
        __syncthreads();
        for (int j = 0; j < 4; j++) {
            int i = j * 128 + t;
            { int r = i >> 4, c4 = i & 15;
              async_ld_b128(baseK + (unsigned)(r * 136 + c4 * 8) * 2,
                  (unsigned long long)(uintptr_t)((const uint4*)kb +
                      (size_t)(h * L_ + kt * 32 + r) * 16 + c4)); }
            { int r = i >> 2, c4 = i & 3;
              async_ld_b128(baseV + (unsigned)(r * 40 + c4 * 8) * 2,
                  (unsigned long long)(uintptr_t)((const uint4*)vT +
                      (size_t)(h * HD_ + r) * 256 + kt * 4 + c4)); }
        }
        WAIT_ASYNC_0();
        __syncthreads();
        v8f s0 = z, s1 = z;
        for (int f = 0; f < 4; f++) {
            bf16x16 bk0 = frag_b_lds(ldsK, 136, 0, f * 32);
            bf16x16 bk1 = frag_b_lds(ldsK, 136, 16, f * 32);
            s0 = wmma_bf16(qf[f], bk0, s0);
            s1 = wmma_bf16(qf[f], bk1, s1);
        }
        // online softmax; C-layout rows: lanes<16 -> M=i, lanes>=16 -> M=8+i
        for (int i = 0; i < 8; i++) {
            float mx = fmaxf(s0[i], s1[i]);
            for (int msk = 8; msk >= 1; msk >>= 1) mx = fmaxf(mx, __shfl_xor(mx, msk, 32));
            float mn = fmaxf(m_[i], mx);
            float alpha = __expf(m_[i] - mn);
            float p0 = __expf(s0[i] - mn);
            float p1 = __expf(s1[i] - mn);
            float rs = p0 + p1;
            for (int msk = 8; msk >= 1; msk >>= 1) rs += __shfl_xor(rs, msk, 32);
            l_[i] = l_[i] * alpha + rs;
            m_[i] = mn;
            for (int n = 0; n < 8; n++) acc[n][i] *= alpha;
            int row = i + 8 * (lane >> 4);
            ldsP[w * 640 + row * 40 + (lane & 15)] = f2bf(p0);
            ldsP[w * 640 + row * 40 + 16 + (lane & 15)] = f2bf(p1);
        }
        // P (16x32) as A-fragment; V^T gives contiguous B-fragments
        bf16x16 pf = frag_a_lds(ldsP + w * 640, 40, 0);
        for (int n = 0; n < 8; n++) {
            bf16x16 bv = frag_b_lds(ldsV, 40, n * 16, 0);
            acc[n] = wmma_bf16(pf, bv, acc[n]);
        }
    }
    float invl[8];
    for (int i = 0; i < 8; i++) invl[i] = 1.0f / l_[i];
    for (int n = 0; n < 8; n++)
        for (int i = 0; i < 8; i++) {
            int row = q0 + w * 16 + i + 8 * (lane >> 4);
            int col = h * HD_ + n * 16 + (lane & 15);
            cat[(size_t)row * CATW_ + col] = f2bf(acc[n][i] * invl[i]);
        }
}

// ---------------- kernel 6: out = x + gate * (cat @ w2^T + b2) ----------------
__global__ __launch_bounds__(256) void gemm2_kernel(const unsigned short* __restrict__ cat,
                                                    const unsigned short* __restrict__ w2bf,
                                                    const float* __restrict__ b2,
                                                    const float* __restrict__ x,
                                                    const float* __restrict__ mod,
                                                    float* __restrict__ out) {
    __shared__ alignas(16) unsigned short ldsA[2 * 128 * 40];
    __shared__ alignas(16) unsigned short ldsB[2 * 128 * 40];
    const int t = threadIdx.x, w = t >> 5, lane = t & 31;
    const int wm = w >> 2, wn = w & 3;
    const int rowBase = blockIdx.y * 128, colBase = blockIdx.x * 128;
    v8f acc[4][2]; v8f z = {};
    for (int a = 0; a < 4; a++) for (int b = 0; b < 2; b++) acc[a][b] = z;
    gemm_core(cat, w2bf, CATW_, rowBase, colBase, ldsA, ldsB, acc);
    const int kh = lane >> 4, nl = lane & 15;
    for (int mi = 0; mi < 4; mi++)
        for (int ni = 0; ni < 2; ni++) {
            int colg = colBase + wn * 32 + ni * 16 + nl;
            float bias = b2[colg];
            float gate = mod[2 * H_ + colg];
            for (int i = 0; i < 8; i++) {
                int rowg = rowBase + wm * 64 + mi * 16 + i + 8 * kh;
                float v = acc[mi][ni][i] + bias;
                out[(size_t)rowg * H_ + colg] = x[(size_t)rowg * H_ + colg] + gate * v;
            }
        }
}

// ---------------- launch ----------------
extern "C" void kernel_launch(void* const* d_in, const int* in_sizes, int n_in,
                              void* d_out, int out_size, void* d_ws, size_t ws_size,
                              hipStream_t stream) {
    const float* x     = (const float*)d_in[0];
    const float* vec   = (const float*)d_in[1];
    const float* pe    = (const float*)d_in[2];
    const float* w_mod = (const float*)d_in[3];
    const float* b_mod = (const float*)d_in[4];
    const float* w1    = (const float*)d_in[5];
    const float* b1    = (const float*)d_in[6];
    const float* w2    = (const float*)d_in[7];
    const float* b2    = (const float*)d_in[8];
    const float* qs    = (const float*)d_in[9];
    const float* ks    = (const float*)d_in[10];
    float* out = (float*)d_out;

    char* ws = (char*)d_ws;
    size_t off = 0;
    auto alloc = [&](size_t bytes) -> void* {
        void* p = ws + off;
        off += (bytes + 255) & ~(size_t)255;
        return p;
    };
    unsigned short* w1bf = (unsigned short*)alloc((size_t)N1_ * H_ * 2);
    unsigned short* w2bf = (unsigned short*)alloc((size_t)H_ * CATW_ * 2);
    unsigned short* xmod = (unsigned short*)alloc((size_t)L_ * H_ * 2);
    float*          qkvb = (float*)alloc((size_t)L_ * QKV_ * 4);
    unsigned short* catb = (unsigned short*)alloc((size_t)L_ * CATW_ * 2);
    unsigned short* qbuf = (unsigned short*)alloc((size_t)NH_ * L_ * HD_ * 2);
    unsigned short* kbuf = (unsigned short*)alloc((size_t)NH_ * L_ * HD_ * 2);
    unsigned short* vTb  = (unsigned short*)alloc((size_t)NH_ * HD_ * L_ * 2);
    float*          modv = (float*)alloc((size_t)3 * H_ * 4);

    cvt_kernel<<<2048, 256, 0, stream>>>(w1, w1bf, N1_ * H_);
    cvt_kernel<<<2048, 256, 0, stream>>>(w2, w2bf, H_ * CATW_);
    mod_kernel<<<3 * H_, 256, 0, stream>>>(vec, w_mod, b_mod, modv);
    ln_kernel<<<L_, 256, 0, stream>>>(x, modv, xmod);
    gemm1_kernel<<<dim3(N1_ / 128, L_ / 128), 256, 0, stream>>>(xmod, w1bf, b1, qkvb, catb);
    rope_kernel<<<NH_ * L_, 128, 0, stream>>>(qkvb, pe, qs, ks, qbuf, kbuf, vTb);
    attn_kernel<<<dim3(L_ / 64, NH_), 128, 0, stream>>>(qbuf, kbuf, vTb, catb);
    gemm2_kernel<<<dim3(H_ / 128, L_ / 128), 256, 0, stream>>>(catb, w2bf, b2, x, modv, out);
}
